// DeepMemoryLevel_5566277616152
// MI455X (gfx1250) — compile-verified, matrix operand order checked by reading, not run on hardware
//
#include <hip/hip_runtime.h>
#include <math.h>

typedef __bf16 bf16;
typedef __attribute__((ext_vector_type(16))) __bf16 v16bf;
typedef __attribute__((ext_vector_type(8)))  __bf16 v8bf;
typedef __attribute__((ext_vector_type(8)))  float  v8f;

// ---- Tensor Data Mover availability / arity detection ----------------------
#if __has_include(<hip/amd_detail/amd_gfx1250_TDM.h>)
#define ATLAS_TDM_6ARG 1
#endif
#if __has_builtin(__builtin_amdgcn_tensor_load_to_lds) && \
    __has_builtin(__builtin_amdgcn_s_wait_tensorcnt)
#define ATLAS_HAVE_TDM 1
#else
#define ATLAS_HAVE_TDM 0
#endif

typedef __attribute__((ext_vector_type(4))) unsigned tdm_u4;
typedef __attribute__((ext_vector_type(8))) int      tdm_i8;
typedef __attribute__((ext_vector_type(4))) int      tdm_i4;

__device__ __forceinline__ bf16 f2bf(float f) {
  unsigned u = __builtin_bit_cast(unsigned, f);
  unsigned r = u + 0x7FFFu + ((u >> 16) & 1u);
  return __builtin_bit_cast(bf16, (unsigned short)(r >> 16));
}
__device__ __forceinline__ float bf2f(bf16 h) {
  unsigned short s = __builtin_bit_cast(unsigned short, h);
  unsigned u = ((unsigned)s) << 16;
  return __builtin_bit_cast(float, u);
}

#if ATLAS_HAVE_TDM
// Issue a TDM 2D tile load: tile_dim = 128 rows x 32 bf16 cols, into LDS rows
// padded to 48 elements (pad_interval: 16 dwords of data, pad_amount: 8 dwords).
// Rows beyond `rows` are OOB for the tensor -> TDM zero-fills them in LDS.
__device__ __forceinline__ void tdm_load_tile(const bf16* gptr, unsigned lds_off,
                                              int rows, int cols_total, int row_stride) {
  unsigned long long ga = (unsigned long long)gptr;
  tdm_u4 g0;
  g0[0] = 1u;                                            // count=1 (valid user D#)
  g0[1] = lds_off;                                       // LDS byte address
  g0[2] = (unsigned)(ga & 0xFFFFFFFFu);                  // global_addr[31:0]
  g0[3] = (unsigned)((ga >> 32) & 0x01FFFFFFu) | (2u << 30); // addr[56:32] | type=2
  unsigned td0 = (unsigned)cols_total;                   // tensor_dim0 (elements)
  unsigned td1 = (unsigned)rows;                         // tensor_dim1
  unsigned long long st = (unsigned long long)row_stride;// tensor_dim0_stride
  tdm_i8 g1;
  g1[0] = (int)((1u << 16)      // data_size = 2 bytes
              | (1u << 20)      // pad_enable
              | (3u << 22)      // pad_interval: 16 DWORDs of data
              | (7u << 25));    // pad_amount:   8 DWORDs of pad
  g1[1] = (int)((td0 & 0xFFFFu) << 16);                  // tensor_dim0[15:0] @bit48
  g1[2] = (int)((td0 >> 16) | ((td1 & 0xFFFFu) << 16));  // dim0 hi | dim1 lo
  g1[3] = (int)((td1 >> 16) | (32u << 16));              // dim1 hi | tile_dim0=32
  g1[4] = (int)(128u);                                   // tile_dim1=128, tile_dim2=0
  g1[5] = (int)(st & 0xFFFFFFFFu);                       // dim0_stride lo
  g1[6] = (int)((st >> 32) & 0xFFFFu);                   // dim0_stride hi | dim1_stride lo=0
  g1[7] = 0;
  tdm_i4 z4 = {0, 0, 0, 0};
#ifdef ATLAS_TDM_6ARG
  tdm_i8 z8 = {0, 0, 0, 0, 0, 0, 0, 0};
  __builtin_amdgcn_tensor_load_to_lds(g0, g1, z4, z4, z8, 0);
#else
  __builtin_amdgcn_tensor_load_to_lds(g0, g1, z4, z4, 0);
#endif
}
#endif

// ---------------------------------------------------------------------------
// WMMA GEMM: C(f32, MxN) = A x B  with bf16 operands, f32 accumulate.
//   AT  == false: A stored row-major (M x K);  AT == true: A stored (K x M)
//   BKN == false: B stored (N x K) row-major (i.e. C = A * Bstored^T)
//   BKN == true : B stored (K x N) row-major
// Block tile 128x128, K-step 32. 8 waves, each wave owns 64x32 (4x2 wmma).
// For the (!AT, !BKN) mode the tiles are staged by the Tensor Data Mover with
// a ping-pong LDS buffer (issue t+1, s_wait_tensorcnt(2) -> tile t ready).
// ---------------------------------------------------------------------------
template<bool AT, bool BKN>
__global__ __launch_bounds__(256)
void gemm_wmma(const bf16* __restrict__ A, const bf16* __restrict__ B,
               float* __restrict__ C, int M, int N, int K)
{
  __shared__ __align__(16) bf16 sA[2][128][48];
  __shared__ __align__(16) bf16 sB[2][128][48];
  const int t    = threadIdx.x;
  const int m0   = blockIdx.y * 128;
  const int n0   = blockIdx.x * 128;
  const int lane = t & 31;
  const int wave = t >> 5;
  const int wm   = wave >> 2;        // 0..1  (M direction)
  const int wn   = wave & 3;         // 0..3  (N direction)
  const int lh   = (lane >> 4) & 1;  // lane half
  const int lr   = lane & 15;

  v8f acc[4][2] = {};

  auto compute_tile = [&](int buf) {
    v16bf afr[4], bfr[2];
    #pragma unroll
    for (int mt = 0; mt < 4; mt++) {
      int ar = wm * 64 + mt * 16 + lr;
      v8bf lo = *(const v8bf*)&sA[buf][ar][lh * 8];
      v8bf hi = *(const v8bf*)&sA[buf][ar][16 + lh * 8];
      afr[mt] = __builtin_shufflevector(lo, hi, 0,1,2,3,4,5,6,7,8,9,10,11,12,13,14,15);
    }
    #pragma unroll
    for (int nt = 0; nt < 2; nt++) {
      int br = wn * 32 + nt * 16 + lr;
      v8bf lo = *(const v8bf*)&sB[buf][br][lh * 8];
      v8bf hi = *(const v8bf*)&sB[buf][br][16 + lh * 8];
      bfr[nt] = __builtin_shufflevector(lo, hi, 0,1,2,3,4,5,6,7,8,9,10,11,12,13,14,15);
    }
    #pragma unroll
    for (int mt = 0; mt < 4; mt++)
      #pragma unroll
      for (int nt = 0; nt < 2; nt++)
        acc[mt][nt] = __builtin_amdgcn_wmma_f32_16x16x32_bf16(
            false, afr[mt], false, bfr[nt], (short)0, acc[mt][nt], false, false);
  };

  auto store_C = [&]() {
    #pragma unroll
    for (int mt = 0; mt < 4; mt++)
      #pragma unroll
      for (int nt = 0; nt < 2; nt++)
        #pragma unroll
        for (int i = 0; i < 8; i++) {
          int row = m0 + wm * 64 + mt * 16 + i + lh * 8;
          int col = n0 + wn * 32 + nt * 16 + lr;
          if (row < M && col < N) C[(size_t)row * N + col] = acc[mt][nt][i];
        }
  };

#if ATLAS_HAVE_TDM
  if constexpr (!AT && !BKN) {
    // ---- Tensor Data Mover pipelined path (dominant-FLOP GEMM mode) -------
    const unsigned ldsA[2] = { (unsigned)(unsigned long long)&sA[0][0][0],
                               (unsigned)(unsigned long long)&sA[1][0][0] };
    const unsigned ldsB[2] = { (unsigned)(unsigned long long)&sB[0][0][0],
                               (unsigned)(unsigned long long)&sB[1][0][0] };
    const int T = K >> 5;   // K is always a multiple of 32 in this workload
    if (wave == 0) {
      tdm_load_tile(A + (size_t)m0 * K, ldsA[0], M - m0, K, K);
      tdm_load_tile(B + (size_t)n0 * K, ldsB[0], N - n0, K, K);
    }
    for (int tt = 0; tt < T; tt++) {
      int cur = tt & 1, nxt = cur ^ 1;
      if (wave == 0) {
        if (tt + 1 < T) {
          tdm_load_tile(A + (size_t)m0 * K + (tt + 1) * 32, ldsA[nxt], M - m0, K, K);
          tdm_load_tile(B + (size_t)n0 * K + (tt + 1) * 32, ldsB[nxt], N - n0, K, K);
          __builtin_amdgcn_s_wait_tensorcnt((short)2);  // tile tt complete
        } else {
          __builtin_amdgcn_s_wait_tensorcnt((short)0);
        }
      }
      __syncthreads();
      compute_tile(cur);
      __syncthreads();      // everyone done reading `cur` before it is refilled
    }
    store_C();
    return;
  }
#endif

  // ---- manual staging path (transpose modes / no-TDM fallback) ------------
  for (int kt = 0; kt < K; kt += 32) {
    __syncthreads();
    // stage A tile (128 x 32) into sA[0][m][k]
    if (!AT) {
      int r = t >> 1, hh = (t & 1) << 4;
      int gm = m0 + r;
      if (gm < M) {
        const uint4* p = (const uint4*)(A + (size_t)gm * K + kt + hh);
        uint4 d0 = p[0]; uint4 d1 = p[1];
        *(uint4*)&sA[0][r][hh]     = d0;
        *(uint4*)&sA[0][r][hh + 8] = d1;
      } else {
        uint4 zz = {0u,0u,0u,0u};
        *(uint4*)&sA[0][r][hh] = zz; *(uint4*)&sA[0][r][hh + 8] = zz;
      }
    } else {
      // A stored (K x M): vector load along M, scalar-transpose into LDS
      int k  = t >> 3;
      int mb = (t & 7) << 4;
      const bf16* src = A + (size_t)(kt + k) * M + m0 + mb;
      v8bf d0 = *(const v8bf*)src;
      v8bf d1 = *(const v8bf*)(src + 8);
      #pragma unroll
      for (int i = 0; i < 8; i++) sA[0][mb + i][k] = d0[i];
      #pragma unroll
      for (int i = 0; i < 8; i++) sA[0][mb + 8 + i][k] = d1[i];
    }
    // stage B tile into sB[0][n][k]
    if (!BKN) {
      int r = t >> 1, hh = (t & 1) << 4;
      int gn = n0 + r;
      if (gn < N) {
        const uint4* p = (const uint4*)(B + (size_t)gn * K + kt + hh);
        uint4 d0 = p[0]; uint4 d1 = p[1];
        *(uint4*)&sB[0][r][hh]     = d0;
        *(uint4*)&sB[0][r][hh + 8] = d1;
      } else {
        uint4 zz = {0u,0u,0u,0u};
        *(uint4*)&sB[0][r][hh] = zz; *(uint4*)&sB[0][r][hh + 8] = zz;
      }
    } else {
      // B stored (K x N): vector load along N, scalar-transpose into LDS
      int k  = t >> 3;
      int nb = (t & 7) << 4;
      const bf16* src = B + (size_t)(kt + k) * N + n0 + nb;
      v8bf d0 = *(const v8bf*)src;
      v8bf d1 = *(const v8bf*)(src + 8);
      #pragma unroll
      for (int i = 0; i < 8; i++) sB[0][nb + i][k] = d0[i];
      #pragma unroll
      for (int i = 0; i < 8; i++) sB[0][nb + 8 + i][k] = d1[i];
    }
    if (kt + 32 < K) {
      int r = t >> 1;
      if (!AT)  { int gm = m0 + r; if (gm < M) __builtin_prefetch(A + (size_t)gm * K + kt + 32, 0, 1); }
      if (!BKN) { int gn = n0 + r; if (gn < N) __builtin_prefetch(B + (size_t)gn * K + kt + 32, 0, 1); }
    }
    __syncthreads();
    compute_tile(0);
  }
  store_C();
}

// ---------------------------------------------------------------------------
// Elementwise / reduction helpers
// ---------------------------------------------------------------------------
__device__ __forceinline__ float block_reduce_sum(float v, float* red) {
  int t = threadIdx.x;
  red[t] = v; __syncthreads();
  for (int s = blockDim.x >> 1; s > 0; s >>= 1) {
    if (t < s) red[t] += red[t + s];
    __syncthreads();
  }
  float r = red[0];
  __syncthreads();
  return r;
}

__global__ void cast_bf_kernel(const float* __restrict__ in, bf16* __restrict__ out, int n) {
  int i = blockIdx.x * blockDim.x + threadIdx.x;
  if (i < n) out[i] = f2bf(in[i]);
}
__global__ void copy_cast_kernel(const float* __restrict__ in, float* __restrict__ f,
                                 bf16* __restrict__ b, int n) {
  int i = blockIdx.x * blockDim.x + threadIdx.x;
  if (i < n) { float v = in[i]; f[i] = v; b[i] = f2bf(v); }
}
__global__ void zero_kernel(float* p, int n) {
  int i = blockIdx.x * blockDim.x + threadIdx.x;
  if (i < n) p[i] = 0.0f;
}

// normalize rows (len Mc) then poly expand: out[:,0:Mc]=x/sqrt2, out[:,Mc:2Mc]=x^2/sqrt2
__global__ void rownorm_poly_kernel(const float* __restrict__ raw, bf16* __restrict__ poly, int Mc) {
  __shared__ float red[256];
  int row = blockIdx.x;
  const float* r = raw + (size_t)row * Mc;
  float s = 0.f;
  for (int j = threadIdx.x; j < Mc; j += blockDim.x) { float v = r[j]; s += v * v; }
  float tot = block_reduce_sum(s, red);
  float inv = 1.0f / fmaxf(sqrtf(tot), 1e-12f);
  const float is2 = 0.70710678118f;
  bf16* p = poly + (size_t)row * (2 * Mc);
  for (int j = threadIdx.x; j < Mc; j += blockDim.x) {
    float v = r[j] * inv;
    p[j]      = f2bf(v * is2);
    p[Mc + j] = f2bf(v * v * is2);
  }
}

__global__ void gates_kernel(const float* __restrict__ x,
    const float* wlr, const float* wmom, const float* wdec, const float* wg,
    const float* blr, const float* bmom, const float* bdec, const float* bg,
    float* lr_row, float* mom_row, float* dec_row, float* og_row, int D) {
  __shared__ float red[256];
  int row = blockIdx.x;
  const float* xr = x + (size_t)row * D;
  float s0 = 0, s1 = 0, s2 = 0, s3 = 0;
  for (int j = threadIdx.x; j < D; j += blockDim.x) {
    float v = xr[j];
    s0 += v * wlr[j]; s1 += v * wmom[j]; s2 += v * wdec[j]; s3 += v * wg[j];
  }
  float t0 = block_reduce_sum(s0, red);
  float t1 = block_reduce_sum(s1, red);
  float t2 = block_reduce_sum(s2, red);
  float t3 = block_reduce_sum(s3, red);
  if (threadIdx.x == 0) {
    lr_row[row]  = 1.f / (1.f + expf(-(t0 + blr[0])));
    mom_row[row] = 1.f / (1.f + expf(-(t1 + bmom[0])));
    dec_row[row] = 1.f / (1.f + expf(-(t2 + bdec[0])));
    og_row[row]  = 1.f / (1.f + expf(-(t3 + bg[0])));
  }
}

__global__ void chunk_mean_kernel(const float* a, const float* b, const float* c,
                                  float* ac, float* bc, float* cc) {
  __shared__ float red[64];
  int ci = blockIdx.x;
  int t = threadIdx.x;                 // 64 threads
  int bb = t >> 5, j = t & 31;
  int row = bb * 1024 + ci * 32 + j;
  float va = a[row], vb = b[row], vc = c[row];
  red[t] = va; __syncthreads();
  for (int s = 32; s > 0; s >>= 1) { if (t < s) red[t] += red[t + s]; __syncthreads(); }
  if (t == 0) ac[ci] = red[0] / 64.f;
  __syncthreads();
  red[t] = vb; __syncthreads();
  for (int s = 32; s > 0; s >>= 1) { if (t < s) red[t] += red[t + s]; __syncthreads(); }
  if (t == 0) bc[ci] = red[0] / 64.f;
  __syncthreads();
  red[t] = vc; __syncthreads();
  for (int s = 32; s > 0; s >>= 1) { if (t < s) red[t] += red[t + s]; __syncthreads(); }
  if (t == 0) cc[ci] = red[0] / 64.f;
}

// gather 64 chunk rows {b*1024 + ci*32 + j} from (2048 x F)
__global__ void gather_rows_bf_kernel(const bf16* __restrict__ src, bf16* __restrict__ dst,
                                      int ci, int F) {
  int idx = blockIdx.x * blockDim.x + threadIdx.x;
  int n = 64 * F;
  if (idx >= n) return;
  int i = idx / F, j = idx - i * F;
  int bb = i >> 5, jj = i & 31;
  dst[idx] = src[(size_t)(bb * 1024 + ci * 32 + jj) * F + j];
}
__global__ void gather_rows_f32_kernel(const float* __restrict__ src, float* __restrict__ dst,
                                       int ci, int F) {
  int idx = blockIdx.x * blockDim.x + threadIdx.x;
  int n = 64 * F;
  if (idx >= n) return;
  int i = idx / F, j = idx - i * F;
  int bb = i >> 5, jj = i & 31;
  dst[idx] = src[(size_t)(bb * 1024 + ci * 32 + jj) * F + j];
}
__global__ void scatter_rows_f32_kernel(const float* __restrict__ src, float* __restrict__ dst,
                                        int ci, int F) {
  int idx = blockIdx.x * blockDim.x + threadIdx.x;
  int n = 64 * F;
  if (idx >= n) return;
  int i = idx / F, j = idx - i * F;
  int bb = i >> 5, jj = i & 31;
  dst[(size_t)(bb * 1024 + ci * 32 + jj) * F + j] = src[idx];
}

__global__ void gelu_kernel(const float* __restrict__ z, float* __restrict__ h,
                            bf16* __restrict__ hb, int n) {
  int i = blockIdx.x * blockDim.x + threadIdx.x;
  if (i >= n) return;
  float v = z[i];
  float g = 0.5f * v * (1.0f + erff(v * 0.70710678118f));
  h[i] = g; hb[i] = f2bf(g);
}
__global__ void e2_kernel(const float* __restrict__ p, const float* __restrict__ v,
                          float* __restrict__ e, bf16* __restrict__ eb, int n) {
  int i = blockIdx.x * blockDim.x + threadIdx.x;
  if (i >= n) return;
  float d = 2.0f * (p[i] - v[i]);
  e[i] = d; eb[i] = f2bf(d);
}
__global__ void dgelu_mul_kernel(const float* __restrict__ z, const float* __restrict__ r,
                                 float* __restrict__ u, bf16* __restrict__ ub, int n) {
  int i = blockIdx.x * blockDim.x + threadIdx.x;
  if (i >= n) return;
  float zz = z[i];
  float cdf = 0.5f * (1.0f + erff(zz * 0.70710678118f));
  float pdf = 0.39894228040f * expf(-0.5f * zz * zz);
  float d = (cdf + zz * pdf) * r[i];
  u[i] = d; ub[i] = f2bf(d);
}

__global__ void row_norm_f32_kernel(const float* __restrict__ m, float* out, int cols) {
  __shared__ float red[256];
  int row = blockIdx.x;
  const float* r = m + (size_t)row * cols;
  float s = 0.f;
  for (int j = threadIdx.x; j < cols; j += blockDim.x) { float v = r[j]; s += v * v; }
  float tot = block_reduce_sum(s, red);
  if (threadIdx.x == 0) out[row] = sqrtf(tot);
}
__global__ void row_norm_bf16_kernel(const bf16* __restrict__ m, float* out, int cols) {
  __shared__ float red[256];
  int row = blockIdx.x;
  const bf16* r = m + (size_t)row * cols;
  float s = 0.f;
  for (int j = threadIdx.x; j < cols; j += blockDim.x) { float v = bf2f(r[j]); s += v * v; }
  float tot = block_reduce_sum(s, red);
  if (threadIdx.x == 0) out[row] = sqrtf(tot);
}

// per-sample clip (||g_i|| = na*nb) folded into rows: out = m / (64*max(na*nb/10,1))
__global__ void scale_rows_kernel(const float* __restrict__ m, const float* na,
                                  const float* nb, bf16* __restrict__ out, int cols) {
  int row = blockIdx.x;
  float nrm = fmaxf(na[row] * nb[row], 1e-8f);
  float sc  = fmaxf(nrm / 10.0f, 1.0f);
  float f   = 1.0f / (64.0f * sc);
  const float* r = m + (size_t)row * cols;
  bf16* o = out + (size_t)row * cols;
  for (int j = threadIdx.x; j < cols; j += blockDim.x) o[j] = f2bf(r[j] * f);
}

__global__ void momentum_kernel(float* __restrict__ Sb, const float* __restrict__ g,
                                const float* momc, const float* lrc, int ci, int n) {
  int i = blockIdx.x * blockDim.x + threadIdx.x;
  if (i < n) Sb[i] = momc[ci] * Sb[i] - lrc[ci] * g[i];
}

__global__ void fro_partial_kernel(const float* __restrict__ g, float* acc, int n) {
  __shared__ float red[256];
  float s = 0.f;
  for (int i = blockIdx.x * blockDim.x + threadIdx.x; i < n; i += gridDim.x * blockDim.x) {
    float v = g[i]; s += v * v;
  }
  float tot = block_reduce_sum(s, red);
  if (threadIdx.x == 0) atomicAdd(acc, tot);
}
__global__ void scale_fro_kernel(const float* __restrict__ g, const float* acc,
                                 float* __restrict__ X, bf16* __restrict__ Xb, int n) {
  float inv = 1.0f / (sqrtf(*acc) + 1e-7f);
  int i = blockIdx.x * blockDim.x + threadIdx.x;
  if (i < n) { float v = g[i] * inv; X[i] = v; Xb[i] = f2bf(v); }
}

__global__ void combine_B_kernel(const float* __restrict__ A, const float* __restrict__ A2,
                                 bf16* __restrict__ Bb, int n) {
  int i = blockIdx.x * blockDim.x + threadIdx.x;
  if (i < n) Bb[i] = f2bf(-4.7750f * A[i] + 2.0315f * A2[i]);
}
__global__ void x_update_kernel(float* __restrict__ X, const float* __restrict__ BX,
                                bf16* __restrict__ Xb, int n) {
  int i = blockIdx.x * blockDim.x + threadIdx.x;
  if (i < n) { float v = 3.4445f * X[i] + BX[i]; X[i] = v; Xb[i] = f2bf(v); }
}
__global__ void weight_update_kernel(float* __restrict__ W, bf16* __restrict__ Wb,
                                     const float* __restrict__ ns, const float* decc,
                                     const float* lrc, int ci, int n) {
  int i = blockIdx.x * blockDim.x + threadIdx.x;
  if (i < n) {
    float v = (1.0f - decc[ci]) * W[i] + ns[i] * lrc[ci];
    W[i] = v; Wb[i] = f2bf(v);
  }
}

__global__ void transpose_f32_kernel(const float* __restrict__ in, float* __restrict__ out,
                                     int R, int Cc) {
  __shared__ float tile[32][33];
  int bx = blockIdx.x * 32, by = blockIdx.y * 32;
  int tx = threadIdx.x, ty = threadIdx.y;   // 32 x 8
  for (int j = ty; j < 32; j += 8) {
    int r = by + j, c = bx + tx;
    if (r < R && c < Cc) tile[j][tx] = in[(size_t)r * Cc + c];
  }
  __syncthreads();
  for (int j = ty; j < 32; j += 8) {
    int r = bx + j, c = by + tx;
    if (r < Cc && c < R) out[(size_t)r * R + c] = tile[tx][j];
  }
}

__global__ void final_out_kernel(const float* __restrict__ x, const float* __restrict__ delta,
                                 const float* __restrict__ og, float* __restrict__ out,
                                 int D, int n) {
  int i = blockIdx.x * blockDim.x + threadIdx.x;
  if (i < n) { int row = i / D; out[i] = x[i] + delta[i] * og[row]; }
}

// ---------------------------------------------------------------------------
// Host orchestration
// ---------------------------------------------------------------------------
extern "C" void kernel_launch(void* const* d_in, const int* in_sizes, int n_in,
                              void* d_out, int out_size, void* d_ws, size_t ws_size,
                              hipStream_t stream) {
  (void)in_sizes; (void)n_in; (void)out_size; (void)ws_size;
  const int D = 2048, M = 512, P = 1024, H = 2048, R = 2048, NC = 32;
  const size_t PH = (size_t)P * H;

  const float* x      = (const float*)d_in[0];
  const float* Wk     = (const float*)d_in[1];
  const float* Wv     = (const float*)d_in[2];
  const float* Wq     = (const float*)d_in[3];
  const float* Wout   = (const float*)d_in[4];
  const float* w_lr   = (const float*)d_in[5];
  const float* b_lr   = (const float*)d_in[6];
  const float* w_mom  = (const float*)d_in[7];
  const float* b_mom  = (const float*)d_in[8];
  const float* w_dec  = (const float*)d_in[9];
  const float* b_dec  = (const float*)d_in[10];
  const float* w_gate = (const float*)d_in[11];
  const float* b_gate = (const float*)d_in[12];
  const float* Wmem0  = (const float*)d_in[13];
  const float* Wmem1  = (const float*)d_in[14];
  const float* Wmemout= (const float*)d_in[15];
  const float* Wvexp  = (const float*)d_in[16];
  float* out = (float*)d_out;

  // bump allocator over d_ws
  char* wsp = (char*)d_ws;
  size_t off = 0;
  auto alloc = [&](size_t bytes) -> void* {
    off = (off + 255) & ~(size_t)255;
    void* p = wsp + off;
    off += bytes;
    return p;
  };
  auto fB = [&](size_t n) { return (float*)alloc(n * 4); };
  auto hB = [&](size_t n) { return (bf16*)alloc(n * 2); };

  bf16 *x_bf = hB((size_t)R * D);
  bf16 *wk_bf = hB((size_t)M * D), *wv_bf = hB((size_t)M * D), *wq_bf = hB((size_t)M * D);
  bf16 *wout_bf = hB((size_t)D * M), *wvexp_bf = hB((size_t)M * P), *wmo_bf = hB((size_t)P * M);
  float *kraw = fB((size_t)R * M), *vraw = fB((size_t)R * M), *qraw = fB((size_t)R * M);
  bf16 *v_bf = hB((size_t)R * M);
  bf16 *kp = hB((size_t)R * P), *qp = hB((size_t)R * P);
  float *vexp = fB((size_t)R * P);
  float *lr_row = fB(R), *mom_row = fB(R), *dec_row = fB(R), *og_row = fB(R);
  float *lr_c = fB(NC), *mom_c = fB(NC), *dec_c = fB(NC);
  float *W0 = fB(PH), *W1 = fB(PH), *S0 = fB(PH), *S1 = fB(PH);
  bf16 *W0b = hB(PH), *W1b = hB(PH);
  float *retrieved = fB((size_t)R * M);
  bf16 *retr_bf = hB((size_t)R * M);
  bf16 *qch = hB(64 * P), *kch = hB(64 * P);
  float *vch = fB(64 * P);
  float *zb = fB(64 * H), *hb = fB(64 * H);
  bf16 *hbb = hB(64 * H);
  float *hy = fB(64 * P);
  bf16 *hyb = hB(64 * P);
  float *retrc = fB(64 * M);
  float *zk = fB(64 * H), *hk = fB(64 * H);
  bf16 *hkb = hB(64 * H);
  float *predk = fB(64 * P), *e2 = fB(64 * P);
  bf16 *e2b = hB(64 * P), *e2s = hB(64 * P);
  float *rbuf = fB(64 * H), *ubuf = fB(64 * H);
  bf16 *usb = hB(64 * H);
  float *nh = fB(64), *ne = fB(64), *nk = fB(64), *nu = fB(64);
  float *g0 = fB(PH), *g1 = fB(PH);
  float *X = fB(PH);
  bf16 *Xb = hB(PH);
  float *Amat = fB((size_t)P * P), *A2 = fB((size_t)P * P);
  bf16 *Abf = hB((size_t)P * P), *Bbf = hB((size_t)P * P);
  float *BX = fB(PH);
  float *fro = fB(1);
  float *delta = fB((size_t)R * D);

  auto el = [&](int n) { return dim3((unsigned)((n + 255) / 256)); };
  auto gNK = [&](const bf16* A, const bf16* B, float* Cc, int m, int n, int k) {
    gemm_wmma<false, false><<<dim3((n + 127) / 128, (m + 127) / 128), 256, 0, stream>>>(A, B, Cc, m, n, k);
  };
  auto gKN = [&](const bf16* A, const bf16* B, float* Cc, int m, int n, int k) {
    gemm_wmma<false, true><<<dim3((n + 127) / 128, (m + 127) / 128), 256, 0, stream>>>(A, B, Cc, m, n, k);
  };
  auto gT = [&](const bf16* A, const bf16* B, float* Cc, int m, int n, int k) {
    gemm_wmma<true, true><<<dim3((n + 127) / 128, (m + 127) / 128), 256, 0, stream>>>(A, B, Cc, m, n, k);
  };
  auto cast = [&](const float* a, bf16* b, int n) {
    cast_bf_kernel<<<el(n), 256, 0, stream>>>(a, b, n);
  };

  // ---- phase A: projections, poly features, gates -------------------------
  cast(x, x_bf, R * D);
  cast(Wk, wk_bf, M * D);  cast(Wv, wv_bf, M * D);  cast(Wq, wq_bf, M * D);
  cast(Wout, wout_bf, D * M);  cast(Wvexp, wvexp_bf, M * P);  cast(Wmemout, wmo_bf, P * M);

  gNK(x_bf, wk_bf, kraw, R, M, D);           // k_raw = x Wk^T
  gNK(x_bf, wv_bf, vraw, R, M, D);
  gNK(x_bf, wq_bf, qraw, R, M, D);
  rownorm_poly_kernel<<<R, 256, 0, stream>>>(kraw, kp, M);
  rownorm_poly_kernel<<<R, 256, 0, stream>>>(qraw, qp, M);
  cast(vraw, v_bf, R * M);
  gKN(v_bf, wvexp_bf, vexp, R, P, M);        // vexp = v Wvexp

  gates_kernel<<<R, 256, 0, stream>>>(x, w_lr, w_mom, w_dec, w_gate,
                                      b_lr, b_mom, b_dec, b_gate,
                                      lr_row, mom_row, dec_row, og_row, D);
  chunk_mean_kernel<<<NC, 64, 0, stream>>>(lr_row, mom_row, dec_row, lr_c, mom_c, dec_c);

  copy_cast_kernel<<<el((int)PH), 256, 0, stream>>>(Wmem0, W0, W0b, (int)PH);
  copy_cast_kernel<<<el((int)PH), 256, 0, stream>>>(Wmem1, W1, W1b, (int)PH);
  zero_kernel<<<el((int)PH), 256, 0, stream>>>(S0, (int)PH);
  zero_kernel<<<el((int)PH), 256, 0, stream>>>(S1, (int)PH);

  auto run_ns5 = [&]() {
    for (int it = 0; it < 5; it++) {
      gNK(Xb, Xb, Amat, P, P, H);                     // A = X X^T     (TDM path)
      cast(Amat, Abf, P * P);
      gNK(Abf, Abf, A2, P, P, P);                     // A^2 (A symm)  (TDM path)
      combine_B_kernel<<<el(P * P), 256, 0, stream>>>(Amat, A2, Bbf, P * P);
      gKN(Bbf, Xb, BX, P, H, P);                      // B X
      x_update_kernel<<<el((int)PH), 256, 0, stream>>>(X, BX, Xb, (int)PH);
    }
  };

  // ---- phase B: sequential chunked ATLAS scan -----------------------------
  for (int ci = 0; ci < NC; ci++) {
    gather_rows_bf_kernel<<<el(64 * P), 256, 0, stream>>>(qp, qch, ci, P);
    gather_rows_bf_kernel<<<el(64 * P), 256, 0, stream>>>(kp, kch, ci, P);
    gather_rows_f32_kernel<<<el(64 * P), 256, 0, stream>>>(vexp, vch, ci, P);

    // retrieve with pre-update params
    gKN(qch, W0b, zb, 64, H, P);
    gelu_kernel<<<el(64 * H), 256, 0, stream>>>(zb, hb, hbb, 64 * H);
    gKN(hbb, W1b, hy, 64, P, H);
    cast(hy, hyb, 64 * P);
    gKN(hyb, wmo_bf, retrc, 64, M, P);
    scatter_rows_f32_kernel<<<el(64 * M), 256, 0, stream>>>(retrc, retrieved, ci, M);

    // per-sample grads (rank-1 factored) + clip + mean
    gKN(kch, W0b, zk, 64, H, P);
    gelu_kernel<<<el(64 * H), 256, 0, stream>>>(zk, hk, hkb, 64 * H);
    gKN(hkb, W1b, predk, 64, P, H);
    e2_kernel<<<el(64 * P), 256, 0, stream>>>(predk, vch, e2, e2b, 64 * P);
    // g1 = hk^T diag(1/(64*clip)) e2
    row_norm_f32_kernel<<<64, 256, 0, stream>>>(hk, nh, H);
    row_norm_f32_kernel<<<64, 256, 0, stream>>>(e2, ne, P);
    scale_rows_kernel<<<64, 256, 0, stream>>>(e2, nh, ne, e2s, P);
    gT(hkb, e2s, g1, H, P, 64);
    // g0 = k^T diag(1/(64*clip)) [gelu'(zk) * (e2 W1^T)]
    gNK(e2b, W1b, rbuf, 64, H, P);
    dgelu_mul_kernel<<<el(64 * H), 256, 0, stream>>>(zk, rbuf, ubuf, usb, 64 * H);
    row_norm_bf16_kernel<<<64, 256, 0, stream>>>(kch, nk, P);
    row_norm_f32_kernel<<<64, 256, 0, stream>>>(ubuf, nu, H);
    scale_rows_kernel<<<64, 256, 0, stream>>>(ubuf, nk, nu, usb, H);
    gT(kch, usb, g0, P, H, 64);

    // surprise momentum
    momentum_kernel<<<el((int)PH), 256, 0, stream>>>(S0, g0, mom_c, lr_c, ci, (int)PH);
    momentum_kernel<<<el((int)PH), 256, 0, stream>>>(S1, g1, mom_c, lr_c, ci, (int)PH);

    // NS5(s0): (P,H) already wide-major
    zero_kernel<<<1, 32, 0, stream>>>(fro, 1);
    fro_partial_kernel<<<512, 256, 0, stream>>>(S0, fro, (int)PH);
    scale_fro_kernel<<<el((int)PH), 256, 0, stream>>>(S0, fro, X, Xb, (int)PH);
    run_ns5();
    weight_update_kernel<<<el((int)PH), 256, 0, stream>>>(W0, W0b, X, dec_c, lr_c, ci, (int)PH);

    // NS5(s1): transpose (H,P) -> (P,H), iterate, transpose back
    zero_kernel<<<1, 32, 0, stream>>>(fro, 1);
    fro_partial_kernel<<<512, 256, 0, stream>>>(S1, fro, (int)PH);
    transpose_f32_kernel<<<dim3(P / 32, H / 32), dim3(32, 8), 0, stream>>>(S1, g0, H, P);
    scale_fro_kernel<<<el((int)PH), 256, 0, stream>>>(g0, fro, X, Xb, (int)PH);
    run_ns5();
    transpose_f32_kernel<<<dim3(H / 32, P / 32), dim3(32, 8), 0, stream>>>(X, g0, P, H);
    weight_update_kernel<<<el((int)PH), 256, 0, stream>>>(W1, W1b, g0, dec_c, lr_c, ci, (int)PH);
  }

  // ---- phase C: output projection + gated residual ------------------------
  cast(retrieved, retr_bf, R * M);
  gNK(retr_bf, wout_bf, delta, R, D, M);     // delta = retrieved Wout^T
  final_out_kernel<<<el(R * D), 256, 0, stream>>>(x, delta, og_row, out, D, R * D);
}
